// CausalAttentionV1_76914274336908
// MI455X (gfx1250) — matrix operand order repaired and checked by reading, not compiled, not run on hardware
//
#include <hip/hip_runtime.h>
#include <hip/hip_bf16.h>
#include <math.h>

// ---------------- CDNA5 WMMA types ----------------
typedef __attribute__((ext_vector_type(16))) __bf16        v16bf;
typedef __attribute__((ext_vector_type(8)))  float         v8f;
typedef __attribute__((ext_vector_type(4)))  unsigned int  v4u;

union BF16Frag { v4u u[2]; v16bf v; };

// Problem sizes (fixed by the reference)
#define BATCH 4
#define SEQ   2048
#define EDIM  1024
#define DDIM  1024
#define MTOT  (BATCH * SEQ)      // 8192

// LDS tile rows are 32 bf16 padded to 40 (80B rows keep 16B alignment)
#define LDS_STRIDE 40

// ---- Workspace layout (bytes). Peak ~134 MB. ----
#define OFF_XB 0u                      // X bf16            16 MiB (dead after QKV gemm)
#define OFF_WT 16777216u               // Wt q,k,v bf16      6 MiB (dead after QKV gemm)
#define OFF_QB 23068672u               // Q  bf16 [8192x1024]   16 MiB
#define OFF_KB 39845888u               // K  bf16 [8192x1024]   16 MiB
#define OFF_VT 56623104u               // Vt bf16 [1024x8192]   16 MiB
#define OFF_SC 73400320u               // scores f32 [4x2048x2048] 64 MiB
#define OFF_PB 0u                      // P bf16 [4x2048x2048] 32 MiB, aliases XB/WT (safe: written after they die)

// ---- gfx1250 async global->LDS staging (ASYNCcnt-tracked), with fallback ----
#if __has_builtin(__builtin_amdgcn_global_load_async_to_lds_b128) && \
    __has_builtin(__builtin_amdgcn_s_wait_asynccnt)
#define USE_ASYNC_LDS 1
#else
#define USE_ASYNC_LDS 0
#endif

#if USE_ASYNC_LDS
// Builtin signature (from hipcc diagnostic): param0 = AS1 pointer to
// int __attribute__((vector_size(16))), param1 = AS3 pointer to same.
typedef int v4i_vs __attribute__((vector_size(16)));
typedef __attribute__((address_space(1))) v4i_vs* as1_v4i;
typedef __attribute__((address_space(3))) v4i_vs* as3_v4i;
#endif

__device__ __forceinline__ void copy16B(const __hip_bfloat16* __restrict__ g,
                                        __hip_bfloat16* l) {
#if USE_ASYNC_LDS
  // global flat address == AS1 numeric address; LDS needs a real addrspacecast
  __builtin_amdgcn_global_load_async_to_lds_b128(
      (as1_v4i)(unsigned long long)g, (as3_v4i)l, 0, 0);
#else
  *(v4u*)l = *(const v4u*)g;
#endif
}

__device__ __forceinline__ void async_wait() {
#if USE_ASYNC_LDS
  __builtin_amdgcn_s_wait_asynccnt(0);
#endif
}

// Stage 256x32 bf16 tile (A) : 1024 16B segments, 4 per thread
__device__ __forceinline__ void stage_A(const __hip_bfloat16* __restrict__ g,
                                        long ld, int kc, __hip_bfloat16* l) {
  const int t = threadIdx.x;
#pragma unroll
  for (int it = 0; it < 4; ++it) {
    int seg = t + it * 256;
    int row = seg >> 2;
    int s8  = (seg & 3) * 8;
    copy16B(g + (long)row * ld + kc + s8, l + row * LDS_STRIDE + s8);
  }
}

// Stage 128x32 bf16 tile (B) : 512 16B segments, 2 per thread
__device__ __forceinline__ void stage_B(const __hip_bfloat16* __restrict__ g,
                                        long ld, int kc, __hip_bfloat16* l) {
  const int t = threadIdx.x;
#pragma unroll
  for (int it = 0; it < 2; ++it) {
    int seg = t + it * 256;
    int row = seg >> 2;
    int s8  = (seg & 3) * 8;
    copy16B(g + (long)row * ld + kc + s8, l + row * LDS_STRIDE + s8);
  }
}

// =====================================================================
// One K=32 step. Each wave owns a 32x128 strip: 2 A fragments (M-subtiles),
// 8 B fragments each reused across both subtiles -> 16 WMMA / 20 ds_loads.
// Fragment layouts per CDNA5 ISA 7.12.2:
//   A (16x32): lanes hi=0 -> k{0..7},{16..23}; hi=1 -> k{8..15},{24..31}
//   B (32x16): lanes hi=0 -> k{0..7},{16..23}; hi=1 -> k{8..15},{24..31}; col = lane&15
// =====================================================================
__device__ __forceinline__ void wmma_step2(const __hip_bfloat16* lA,
                                           const __hip_bfloat16* lB,
                                           v8f acc[2][8]) {
  const int lane = threadIdx.x & 31;
  const int hi   = lane >> 4;
  const int m    = lane & 15;
  const int wave = threadIdx.x >> 5;

  BF16Frag a0, a1;
  const __hip_bfloat16* ar0 = lA + (wave * 32 + m) * LDS_STRIDE + hi * 8;
  const __hip_bfloat16* ar1 = ar0 + 16 * LDS_STRIDE;
  a0.u[0] = *(const v4u*)(ar0);
  a0.u[1] = *(const v4u*)(ar0 + 16);
  a1.u[0] = *(const v4u*)(ar1);
  a1.u[1] = *(const v4u*)(ar1 + 16);

#pragma unroll
  for (int j = 0; j < 8; ++j) {
    BF16Frag b;
    const __hip_bfloat16* br = lB + (j * 16 + m) * LDS_STRIDE + hi * 8;
    b.u[0] = *(const v4u*)(br);
    b.u[1] = *(const v4u*)(br + 16);
    acc[0][j] = __builtin_amdgcn_wmma_f32_16x16x32_bf16(
        false, a0.v, false, b.v, (short)0, acc[0][j], false, false);
    acc[1][j] = __builtin_amdgcn_wmma_f32_16x16x32_bf16(
        false, a1.v, false, b.v, (short)0, acc[1][j], false, false);
  }
}

// =====================================================================
// Shared GEMM core: C[256 x 128] += A[256 x kend] * B[128 x kend]^T
// (both operands row-major with contiguous K). Double-buffered LDS,
// one barrier per K-step, async staging overlapped with WMMA.
// =====================================================================
__device__ __forceinline__ void gemm_core(const __hip_bfloat16* __restrict__ A, long lda,
                                          const __hip_bfloat16* __restrict__ B, long ldb,
                                          int kend, v8f acc[2][8]) {
  __shared__ __align__(16) __hip_bfloat16 lA[2][256 * LDS_STRIDE];
  __shared__ __align__(16) __hip_bfloat16 lB[2][128 * LDS_STRIDE];

  stage_A(A, lda, 0, lA[0]);
  stage_B(B, ldb, 0, lB[0]);
  async_wait();
  __syncthreads();

  int p = 0;
  for (int kc = 0; kc < kend; kc += 32) {
    const int kn = kc + 32;
    if (kn < kend) {                    // prefetch next tile into other buffer
      stage_A(A, lda, kn, lA[p ^ 1]);
      stage_B(B, ldb, kn, lB[p ^ 1]);
    }
    wmma_step2(lA[p], lB[p], acc);
    async_wait();
    __syncthreads();
    p ^= 1;
  }
}

__device__ __forceinline__ void zero_acc(v8f acc[2][8]) {
#pragma unroll
  for (int s = 0; s < 2; ++s)
#pragma unroll
    for (int j = 0; j < 8; ++j) acc[s][j] = (v8f){0, 0, 0, 0, 0, 0, 0, 0};
}

// =====================================================================
// Conversion kernels
// =====================================================================
__global__ void conv_x_kernel(const float* __restrict__ X,
                              __hip_bfloat16* __restrict__ XB, int n) {
  for (int i = blockIdx.x * 256 + threadIdx.x; i < n; i += gridDim.x * 256)
    XB[i] = __float2bfloat16(X[i]);
}

// Wt[z][n][k] = bf16(W[z][k][n])  (transpose so B operand rows are contiguous-K)
__global__ void conv_w_kernel(const float* __restrict__ Wq,
                              const float* __restrict__ Wk,
                              const float* __restrict__ Wv,
                              __hip_bfloat16* __restrict__ WT) {
  const int z = blockIdx.z;
  const float* W = (z == 0) ? Wq : (z == 1) ? Wk : Wv;
  __hip_bfloat16* Wt = WT + (long)z * EDIM * DDIM;
  for (int i = blockIdx.x * 256 + threadIdx.x; i < EDIM * DDIM;
       i += gridDim.x * 256) {
    int nn = i & (DDIM - 1);
    int kk = i >> 10;
    Wt[(long)nn * EDIM + kk] = __float2bfloat16(W[i]);
  }
}

// =====================================================================
// QKV GEMM: [8192 x 1024] = XB[8192x1024] * Wt[z]^T   (z = Q,K,V)
// Q,K stored row-major bf16; V stored transposed (Vt[d][gm]).
// =====================================================================
__global__ void qkv_gemm_kernel(const __hip_bfloat16* __restrict__ XB,
                                const __hip_bfloat16* __restrict__ WT,
                                __hip_bfloat16* __restrict__ QB,
                                __hip_bfloat16* __restrict__ KB,
                                __hip_bfloat16* __restrict__ VT) {
  const int z = blockIdx.z;
  const __hip_bfloat16* A = XB + (long)blockIdx.y * 256 * EDIM;
  const __hip_bfloat16* B = WT + (long)z * EDIM * DDIM + (long)blockIdx.x * 128 * EDIM;

  v8f acc[2][8];
  zero_acc(acc);
  gemm_core(A, EDIM, B, EDIM, EDIM, acc);

  const int lane = threadIdx.x & 31;
  const int hi   = lane >> 4;
  const int m    = lane & 15;
  const int wave = threadIdx.x >> 5;
  const int colbase = blockIdx.x * 128;

  if (z < 2) {
    __hip_bfloat16* O = z ? KB : QB;
#pragma unroll
    for (int s = 0; s < 2; ++s) {
      const int rowbase = blockIdx.y * 256 + wave * 32 + s * 16 + hi * 8;
#pragma unroll
      for (int j = 0; j < 8; ++j)
#pragma unroll
        for (int r = 0; r < 8; ++r)
          O[(long)(rowbase + r) * DDIM + colbase + j * 16 + m] =
              __float2bfloat16(acc[s][j][r]);
    }
  } else {
#pragma unroll
    for (int s = 0; s < 2; ++s) {
      const int rowbase = blockIdx.y * 256 + wave * 32 + s * 16 + hi * 8;
#pragma unroll
      for (int j = 0; j < 8; ++j)
#pragma unroll
        for (int r = 0; r < 8; ++r)
          VT[(long)(colbase + j * 16 + m) * MTOT + rowbase + r] =
              __float2bfloat16(acc[s][j][r]);
    }
  }
}

// =====================================================================
// Scores GEMM per batch: SC[q][key] = (Q_b * K_b^T) / 32 ; causal block skip
// =====================================================================
__global__ void scores_gemm_kernel(const __hip_bfloat16* __restrict__ QB,
                                   const __hip_bfloat16* __restrict__ KB,
                                   float* __restrict__ SC) {
  // tile cols start at bx*128, tile rows end at by*256+255:
  // strictly above diagonal (never read by softmax) iff bx*128 > by*256+255
  if (blockIdx.x >= 2 * blockIdx.y + 2) return;

  const int b = blockIdx.z;
  const __hip_bfloat16* A = QB + (long)b * SEQ * DDIM + (long)blockIdx.y * 256 * DDIM;
  const __hip_bfloat16* B = KB + (long)b * SEQ * DDIM + (long)blockIdx.x * 128 * DDIM;
  float* Sb = SC + (long)b * SEQ * SEQ;

  v8f acc[2][8];
  zero_acc(acc);
  gemm_core(A, DDIM, B, DDIM, DDIM, acc);

  const int lane = threadIdx.x & 31;
  const int hi   = lane >> 4;
  const int m    = lane & 15;
  const int wave = threadIdx.x >> 5;
  const int colbase = blockIdx.x * 128;
  const float scale = 0.03125f;          // 1/sqrt(1024)

#pragma unroll
  for (int s = 0; s < 2; ++s) {
    const int rowbase = blockIdx.y * 256 + wave * 32 + s * 16 + hi * 8;
#pragma unroll
    for (int j = 0; j < 8; ++j)
#pragma unroll
      for (int r = 0; r < 8; ++r)
        Sb[(long)(rowbase + r) * SEQ + colbase + j * 16 + m] =
            acc[s][j][r] * scale;
  }
}

// =====================================================================
// Causal row softmax: one block per query row; writes bf16 P (zero tail)
// =====================================================================
__global__ void softmax_kernel(const float* __restrict__ SC,
                               __hip_bfloat16* __restrict__ PB) {
  const int q = blockIdx.x;
  const int b = blockIdx.y;
  const float* row = SC + ((long)b * SEQ + q) * SEQ;
  __hip_bfloat16* prow = PB + ((long)b * SEQ + q) * SEQ;
  const int n = q + 1;

  __shared__ float red[256];
  const int t = threadIdx.x;

  float mx = -INFINITY;
  for (int j = t; j < n; j += 256) mx = fmaxf(mx, row[j]);
  red[t] = mx;
  __syncthreads();
  for (int s = 128; s > 0; s >>= 1) {
    if (t < s) red[t] = fmaxf(red[t], red[t + s]);
    __syncthreads();
  }
  mx = red[0];
  __syncthreads();

  float sum = 0.0f;
  for (int j = t; j < n; j += 256) sum += __expf(row[j] - mx);
  red[t] = sum;
  __syncthreads();
  for (int s = 128; s > 0; s >>= 1) {
    if (t < s) red[t] += red[t + s];
    __syncthreads();
  }
  const float inv = 1.0f / red[0];

  for (int j = t; j < SEQ; j += 256) {
    float p = (j < n) ? __expf(row[j] - mx) * inv : 0.0f;
    prow[j] = __float2bfloat16(p);
  }
}

// =====================================================================
// Context GEMM per batch: OUT[q][d] = P_b[2048x2048] * Vt_b^T, f32 out.
// K-loop clipped at causal boundary (P is zero beyond, but skip the work).
// =====================================================================
__global__ void ctx_gemm_kernel(const __hip_bfloat16* __restrict__ PB,
                                const __hip_bfloat16* __restrict__ VT,
                                float* __restrict__ OUT) {
  const int b = blockIdx.z;
  const __hip_bfloat16* A = PB + (long)b * SEQ * SEQ + (long)blockIdx.y * 256 * SEQ;
  const __hip_bfloat16* B = VT + (long)b * SEQ + (long)blockIdx.x * 128 * MTOT;
  float* Ob = OUT + (long)b * SEQ * DDIM;

  const int kend = (blockIdx.y + 1) * 256;   // causal: keys <= last query of tile

  v8f acc[2][8];
  zero_acc(acc);
  gemm_core(A, SEQ, B, MTOT, kend, acc);

  const int lane = threadIdx.x & 31;
  const int hi   = lane >> 4;
  const int m    = lane & 15;
  const int wave = threadIdx.x >> 5;
  const int colbase = blockIdx.x * 128;

#pragma unroll
  for (int s = 0; s < 2; ++s) {
    const int rowbase = blockIdx.y * 256 + wave * 32 + s * 16 + hi * 8;
#pragma unroll
    for (int j = 0; j < 8; ++j)
#pragma unroll
      for (int r = 0; r < 8; ++r)
        Ob[(long)(rowbase + r) * DDIM + colbase + j * 16 + m] = acc[s][j][r];
  }
}

// =====================================================================
extern "C" void kernel_launch(void* const* d_in, const int* in_sizes, int n_in,
                              void* d_out, int out_size, void* d_ws, size_t ws_size,
                              hipStream_t stream) {
  const float* X  = (const float*)d_in[0];
  const float* Wq = (const float*)d_in[1];
  const float* Wk = (const float*)d_in[2];
  const float* Wv = (const float*)d_in[3];

  char* ws = (char*)d_ws;
  __hip_bfloat16* XB = (__hip_bfloat16*)(ws + OFF_XB);
  __hip_bfloat16* WT = (__hip_bfloat16*)(ws + OFF_WT);
  __hip_bfloat16* QB = (__hip_bfloat16*)(ws + OFF_QB);
  __hip_bfloat16* KB = (__hip_bfloat16*)(ws + OFF_KB);
  __hip_bfloat16* VT = (__hip_bfloat16*)(ws + OFF_VT);
  float*          SC = (float*)(ws + OFF_SC);
  __hip_bfloat16* PB = (__hip_bfloat16*)(ws + OFF_PB);

  // 1) precision conversion (bf16 is the MI455X matrix-core fast path)
  conv_x_kernel<<<4096, 256, 0, stream>>>(X, XB, MTOT * EDIM);
  conv_w_kernel<<<dim3(1024, 1, 3), 256, 0, stream>>>(Wq, Wk, Wv, WT);

  // 2) QKV projections (grid.z selects Q/K/V; V stored transposed)
  qkv_gemm_kernel<<<dim3(DDIM / 128, MTOT / 256, 3), 256, 0, stream>>>(XB, WT, QB, KB, VT);

  // 3) causal-skipped scores
  scores_gemm_kernel<<<dim3(SEQ / 128, SEQ / 256, BATCH), 256, 0, stream>>>(QB, KB, SC);

  // 4) row softmax -> bf16 P
  softmax_kernel<<<dim3(SEQ, BATCH), 256, 0, stream>>>(SC, PB);

  // 5) context = P * V
  ctx_gemm_kernel<<<dim3(DDIM / 128, SEQ / 256, BATCH), 256, 0, stream>>>(PB, VT, (float*)d_out);
}
